// GRSNN_2199023256073
// MI455X (gfx1250) — matrix operand order repaired and checked by reading, not compiled, not run on hardware
//
#include <hip/hip_runtime.h>
#include <cmath>

#define N_NODE 20000
#define E_EDGE 320000
#define B_Q 4
#define K_T 32
#define D_DIM 32
#define DR_REL 40
#define T_STEPS 10
#define VTH 2.0f
#define NBD (N_NODE * B_Q * D_DIM)          // 2,560,000 floats per state array
#define WLUT_SZ (DR_REL * B_Q * D_DIM)      // 5,120 floats (20 KB)

typedef __attribute__((ext_vector_type(16))) _Float16 v16h;
typedef __attribute__((ext_vector_type(8)))  float    v8f;

// Fragment-major permuted index for (row = node*B+b, d) -> WMMA D-layout order.
// Per 16-row tile: lane = (d&15) + 16*(rowInTile>=8), then frag (d/16) * 8 + rowInTile%8.
__device__ __forceinline__ int frag_idx(int row, int d) {
    int tile = row >> 4, mt = row & 15;
    int lane = (d & 15) + ((mt >> 3) << 4);
    return tile * 512 + lane * 16 + ((d >> 4) << 3) + (mt & 7);
}

// ---------------------------------------------------------------- zero
__global__ void zero_f32(float* __restrict__ p, int n) {
    int stride = gridDim.x * blockDim.x;
    int n4 = n >> 2;
    for (int j = blockIdx.x * blockDim.x + threadIdx.x; j < n4; j += stride)
        ((float4*)p)[j] = make_float4(0.f, 0.f, 0.f, 0.f);
}

// ---------------------------------------------------------------- query gather: Q[b][d] = query_emb[r_index[b]][d]
__global__ void query_kernel(const float* __restrict__ query_emb,
                             const int* __restrict__ r_index,
                             float* __restrict__ Q) {
    int t = threadIdx.x;                      // 128 threads
    int b = t >> 5, d = t & 31;
    Q[t] = query_emb[r_index[b] * D_DIM + d];
}

// ---------------------------------------------------------------- relation weight LUT: WLUT[r][b][d] = (Q @ relw_W + b)[b][r*32+d]
__global__ void wlut_kernel(const float* __restrict__ Q,
                            const float* __restrict__ relw_W,   // (32, 1280)
                            const float* __restrict__ relw_b,   // (1280,)
                            float* __restrict__ WLUT) {
    int idx = blockIdx.x * blockDim.x + threadIdx.x;  // 5120 threads
    if (idx >= WLUT_SZ) return;
    int r = idx >> 7;            // /128
    int b = (idx >> 5) & 3;
    int d = idx & 31;
    float acc = relw_b[r * D_DIM + d];
    const float* w = relw_W + r * D_DIM + d;          // stride DR_REL*D_DIM = 1280
    const float* q = Q + b * D_DIM;
    #pragma unroll 8
    for (int k = 0; k < D_DIM; ++k) acc = fmaf(q[k], w[k * (DR_REL * D_DIM)], acc);
    WLUT[idx] = acc;
}

// ---------------------------------------------------------------- boundary / initial LIF state ((h_index[b], b) pairs are unique)
__global__ void boundary_kernel(const float* __restrict__ Q,
                                const int* __restrict__ h_index,
                                float* __restrict__ S, float* __restrict__ V,
                                float* __restrict__ ENC, float kern0) {
    int t = threadIdx.x;                      // 128 threads
    int b = t >> 5, d = t & 31;
    float qs = Q[t] * (VTH * 0.5f) + VTH * 0.5f;
    int row = h_index[b] * B_Q + b;
    float s0 = (qs - VTH >= 0.f) ? 1.f : 0.f;
    S[row * D_DIM + d] = s0;                  // standard layout (edge gather)
    int fi = frag_idx(row, d);                // fragment layout (node kernel RMW)
    V[fi]   = qs * (1.f - s0);
    ENC[fi] = kern0 * s0;
}

// ---------------------------------------------------------------- edge scatter: wave-per-edge, LDS-resident relation LUT,
//                                                                  spike-gated native f32 atomics (spikes are 0/1 and sparse)
__global__ void edge_kernel(const int* __restrict__ esrc, const int* __restrict__ edst,
                            const int* __restrict__ etyp, const float* __restrict__ wlut,
                            const float* __restrict__ S, float* __restrict__ AGG) {
    __shared__ float wl[WLUT_SZ];             // 20 KB, reused by every edge of the block
    for (int i = threadIdx.x; i < WLUT_SZ; i += blockDim.x) wl[i] = wlut[i];
    __syncthreads();
    const int lane   = threadIdx.x & 31;
    const int nWaves = gridDim.x * (blockDim.x >> 5);
    int w = blockIdx.x * (blockDim.x >> 5) + (threadIdx.x >> 5);
    for (int e = w; e < E_EDGE; e += nWaves) {
        __builtin_prefetch(esrc + e + nWaves, 0, 1);   // global_prefetch_b8
        int src = esrc[e], dst = edst[e], typ = etyp[e];
        const float4 sv = *(const float4*)(S + src * (B_Q * D_DIM) + lane * 4);
        if (sv.x == 0.f && sv.y == 0.f && sv.z == 0.f && sv.w == 0.f) continue;
        const float4 wv = *(const float4*)(wl + typ * (B_Q * D_DIM) + lane * 4);
        float* ag = AGG + dst * (B_Q * D_DIM) + lane * 4;
        // discarded result -> non-returning global_atomic_add_f32
        if (sv.x != 0.f) (void)__hip_atomic_fetch_add(ag + 0, wv.x, __ATOMIC_RELAXED, __HIP_MEMORY_SCOPE_AGENT);
        if (sv.y != 0.f) (void)__hip_atomic_fetch_add(ag + 1, wv.y, __ATOMIC_RELAXED, __HIP_MEMORY_SCOPE_AGENT);
        if (sv.z != 0.f) (void)__hip_atomic_fetch_add(ag + 2, wv.z, __ATOMIC_RELAXED, __HIP_MEMORY_SCOPE_AGENT);
        if (sv.w != 0.f) (void)__hip_atomic_fetch_add(ag + 3, wv.w, __ATOMIC_RELAXED, __HIP_MEMORY_SCOPE_AGENT);
    }
}

// ---------------------------------------------------------------- node kernel: WMMA linear (16 rows x K=32 x 32 cols) + LIF + enc
// wave-per-tile, tiles = N*B/16 = 5000 exactly (EXEC all-ones for WMMA).
// Cs/Vm/ENC live in fragment-major layout -> per-lane contiguous b128 state RMW.
// AGG is consumed exactly once per element here, so we store zeros back for the next step.
__global__ void node_kernel(float* __restrict__ AGG,
                            const float* __restrict__ linW, const float* __restrict__ linb,
                            float* __restrict__ S, float* __restrict__ Vm,
                            float* __restrict__ Cs, float* __restrict__ ENC,
                            float decay, float kern) {
    const int lane = threadIdx.x & 31;
    const int tile = blockIdx.x * (blockDim.x >> 5) + (threadIdx.x >> 5);
    if (tile >= (N_NODE * B_Q) / 16) return;
    const int m  = lane & 15;     // A-row for loads, D-col for stores
    const int hi = lane >> 4;

    // A fragment: 16-bit A 16x32, lane = M, K-halves split by lane group
    v16h a;
    float* ap = AGG + (tile * 16 + m) * D_DIM + hi * 8;
    #pragma unroll
    for (int i = 0; i < 8; ++i) a[i]     = (_Float16)ap[i];        // K = 8*hi + i
    #pragma unroll
    for (int i = 0; i < 8; ++i) a[8 + i] = (_Float16)ap[16 + i];   // K = 16 + 8*hi + i
    // reset AGG in place for the next step's edge scatter (lanes tile the block exactly)
    const float4 z4 = make_float4(0.f, 0.f, 0.f, 0.f);
    ((float4*)ap)[0] = z4; ((float4*)ap)[1] = z4;
    ((float4*)(ap + 16))[0] = z4; ((float4*)(ap + 16))[1] = z4;

    // B fragments: B[k][n] = linW[n][k]  (x = agg @ linW.T); lane = N, K split by lane group
    v16h b0, b1;
    const float* bp0 = linW + m * D_DIM + hi * 8;
    const float* bp1 = linW + (m + 16) * D_DIM + hi * 8;
    #pragma unroll
    for (int i = 0; i < 16; ++i) { int k = (i < 8) ? i : i + 8; b0[i] = (_Float16)bp0[k]; b1[i] = (_Float16)bp1[k]; }

    v8f c0 = {}, c1 = {};
    c0 = __builtin_amdgcn_wmma_f32_16x16x32_f16(false, a, false, b0, (short)0, c0, false, false);
    c1 = __builtin_amdgcn_wmma_f32_16x16x32_f16(false, a, false, b1, (short)0, c1, false, false);

    // x[i]: i = frag*8 + v  (frag0 col = m, frag1 col = m+16; row = tile*16 + v + 8*hi)
    const float bias0 = linb[m], bias1 = linb[m + 16];
    float x[16];
    #pragma unroll
    for (int v = 0; v < 8; ++v) { x[v] = c0[v] + bias0; x[8 + v] = c1[v] + bias1; }

    const int base = tile * 512 + lane * 16;  // fragment-major state base (16 contiguous floats)
    float* csP  = Cs  + base;
    float* vmP  = Vm  + base;
    float* encP = ENC + base;
    #pragma unroll
    for (int i = 0; i < 16; ++i) {
        float cn = csP[i] * decay + x[i];
        float vn = vmP[i] * decay + cn;
        float sn = (vn - VTH >= 0.f) ? 1.f : 0.f;
        vn *= (1.f - sn);
        csP[i] = cn;
        vmP[i] = vn;
        encP[i] += kern * sn;
        // spike scatter to standard (row, d) layout for the edge gather
        int v = i & 7, f = i >> 3;
        S[(tile * 16 + v + 8 * hi) * D_DIM + m + 16 * f] = sn;
    }
}

// ---------------------------------------------------------------- final scorer: gather feat (128 x 64), WMMA MLP, dot with W2
__global__ void final_kernel(const float* __restrict__ ENC, const float* __restrict__ Q,
                             const int* __restrict__ t_index,
                             const float* __restrict__ W1, const float* __restrict__ b1,
                             const float* __restrict__ W2, const float* __restrict__ b2,
                             float* __restrict__ out) {
    __shared__ _Float16 featLds[128 * 64];    // 16 KB
    __shared__ float    hLds[128 * 64];       // 32 KB

    // stage feat = [enc(t_index) | query] as f16 (ENC is fragment-major -> permuted gather)
    for (int idx = threadIdx.x; idx < 128 * 64; idx += blockDim.x) {
        int row = idx >> 6, kk = idx & 63;
        int b = row >> 5, k = row & 31;       // row = b*K + k
        float v = (kk < D_DIM)
                ? ENC[frag_idx(t_index[b * K_T + k] * B_Q + b, kk)]
                : Q[b * D_DIM + (kk - D_DIM)];
        featLds[idx] = (_Float16)v;
    }
    __syncthreads();

    const int w = threadIdx.x >> 5;           // 8 waves -> 8 row tiles
    const int lane = threadIdx.x & 31;
    const int m = lane & 15, hi = lane >> 4;
    const int rowA = w * 16 + m;

    v16h a0, a1;                              // K = 64 in two chunks of 32
    #pragma unroll
    for (int i = 0; i < 16; ++i) {
        int k = 16 * hi + i;
        a0[i] = featLds[rowA * 64 + k];
        a1[i] = featLds[rowA * 64 + 32 + k];
    }

    #pragma unroll
    for (int ct = 0; ct < 4; ++ct) {          // 4 col tiles of 16
        int col = ct * 16 + m;
        v16h bb0, bb1;                        // B[k][n] = W1[n][k] (h = feat @ W1.T)
        const float* wp = W1 + col * 64 + 16 * hi;
        #pragma unroll
        for (int i = 0; i < 16; ++i) { bb0[i] = (_Float16)wp[i]; bb1[i] = (_Float16)wp[32 + i]; }
        v8f c = {};
        c = __builtin_amdgcn_wmma_f32_16x16x32_f16(false, a0, false, bb0, (short)0, c, false, false);
        c = __builtin_amdgcn_wmma_f32_16x16x32_f16(false, a1, false, bb1, (short)0, c, false, false);
        #pragma unroll
        for (int v = 0; v < 8; ++v) {
            int rr = w * 16 + v + 8 * hi;
            hLds[rr * 64 + col] = fmaxf(c[v] + b1[col], 0.f);
        }
    }
    __syncthreads();

    if (threadIdx.x < 128) {
        float acc = b2[0];
        const float* h = hLds + threadIdx.x * 64;
        #pragma unroll 8
        for (int kk = 0; kk < 64; ++kk) acc = fmaf(h[kk], W2[kk], acc);
        out[threadIdx.x] = acc;               // row = b*K + k  ==  (B,K) row-major
    }
}

// ---------------------------------------------------------------- host
extern "C" void kernel_launch(void* const* d_in, const int* in_sizes, int n_in,
                              void* d_out, int out_size, void* d_ws, size_t ws_size,
                              hipStream_t stream) {
    const int*   edge_src  = (const int*)  d_in[0];
    const int*   edge_dst  = (const int*)  d_in[1];
    const int*   edge_type = (const int*)  d_in[2];
    const int*   h_index   = (const int*)  d_in[3];
    const int*   t_index   = (const int*)  d_in[4];
    const int*   r_index   = (const int*)  d_in[5];
    const float* query_emb = (const float*)d_in[6];
    const float* relw_W    = (const float*)d_in[7];
    const float* relw_b    = (const float*)d_in[8];
    const float* lin_W     = (const float*)d_in[9];
    const float* lin_b     = (const float*)d_in[10];
    const float* mlp_W1    = (const float*)d_in[11];
    const float* mlp_b1    = (const float*)d_in[12];
    const float* mlp_W2    = (const float*)d_in[13];
    const float* mlp_b2    = (const float*)d_in[14];

    float* ws   = (float*)d_ws;
    float* S    = ws;                 // (N,B,D) standard layout
    float* V    = S   + NBD;          // fragment-major
    float* C    = V   + NBD;          // fragment-major
    float* ENC  = C   + NBD;          // fragment-major
    float* AGG  = ENC + NBD;          // (N,B,D) standard layout
    float* Q    = AGG + NBD;          // (B,D)
    float* WLUT = Q   + B_Q * D_DIM;  // (DR,B,D)

    const float rr     = 0.95f;
    const float nscale = (1.0f - powf(rr, (float)T_STEPS)) / (1.0f - rr);
    const float decay  = expf(-0.25f);

    zero_f32<<<512, 256, 0, stream>>>(S, 5 * NBD);                 // S,V,C,ENC,AGG contiguous
    query_kernel<<<1, 128, 0, stream>>>(query_emb, r_index, Q);
    wlut_kernel<<<20, 256, 0, stream>>>(Q, relw_W, relw_b, WLUT);
    boundary_kernel<<<1, 128, 0, stream>>>(Q, h_index, S, V, ENC, 1.0f / nscale);

    for (int t = 1; t < T_STEPS; ++t) {
        edge_kernel<<<512, 256, 0, stream>>>(edge_src, edge_dst, edge_type, WLUT, S, AGG);
        node_kernel<<<625, 256, 0, stream>>>(AGG, lin_W, lin_b, S, V, C, ENC,   // node zeroes AGG
                                             decay, powf(rr, (float)t) / nscale);
    }
    final_kernel<<<1, 256, 0, stream>>>(ENC, Q, t_index, mlp_W1, mlp_b1, mlp_W2, mlp_b2,
                                        (float*)d_out);
}
